// JiuZhouBianMa_26079041421868
// MI455X (gfx1250) — matrix operand order repaired and checked by reading, not compiled, not run on hardware
//
#include <hip/hip_runtime.h>
#include <hip/hip_bf16.h>
#include <cstdint>
#include <cstddef>

// Problem constants (from reference): H=8, B=4, S=4096, D=512, E=D+2=514
#define Hh    8
#define Bbat  4
#define Ss    4096
#define Dd    512
#define Ee    514
#define MH    (Bbat * Ss)       // 16384 rows per head
#define M_BLK 64
#define N_BLK 128
#define K_BLK 32
#define A_LD  520               // padded halfword stride of staged A rows (512 + 8)
#define B_LD  40                // padded halfword stride of staged B rows (32 + 8)

typedef __attribute__((ext_vector_type(16))) __bf16 v16bf;
typedef __attribute__((ext_vector_type(8)))  __bf16 v8bf;
typedef __attribute__((ext_vector_type(4)))  __bf16 v4bf;
typedef __attribute__((ext_vector_type(8)))  float  v8f;

struct bfpair { __bf16 hi; __bf16 lo; };

static __device__ __forceinline__ bfpair split2(float x) {
    bfpair r;
    r.hi = (__bf16)x;                   // RTNE to bf16
    r.lo = (__bf16)(x - (float)r.hi);   // residual, |lo| <= 2^-8 |x|
    return r;
}

static __device__ __forceinline__ v16bf cat8(v8bf a, v8bf b) {
    return __builtin_shufflevector(a, b, 0, 1, 2, 3, 4, 5, 6, 7,
                                          8, 9, 10, 11, 12, 13, 14, 15);
}

__global__ __launch_bounds__(256)
void JiuZhou_fused_gemm_gate(const float* __restrict__ heads,
                             const float* __restrict__ Wz,
                             const float* __restrict__ Wg,
                             const float* __restrict__ bg,
                             float* __restrict__ out) {
    extern __shared__ char smem[];
    __bf16* Ahi  = (__bf16*)smem;                 // [M_BLK][A_LD]
    __bf16* Alo  = Ahi + M_BLK * A_LD;            // [M_BLK][A_LD]
    __bf16* Bhi  = Alo + M_BLK * A_LD;            // [N_BLK][B_LD]
    __bf16* Blo  = Bhi + N_BLK * B_LD;            // [N_BLK][B_LD]
    float*  gpart = (float*)(Blo + N_BLK * B_LD); // [M_BLK][4]
    float*  gval  = gpart + M_BLK * 4;            // [M_BLK]

    const int t      = threadIdx.x;
    const int nBlock = blockIdx.x;   // 0..3   (N tiles of 128 over D=512)
    const int mBlock = blockIdx.y;   // 0..255 (M tiles of 64 over 16384)
    const int h      = blockIdx.z;   // 0..7

    // ---------------- Stage A: 64x512 fp32 -> bf16 hi/lo in LDS ----------------
    const float4* hsrc =
        (const float4*)(heads + ((size_t)h * MH + (size_t)mBlock * M_BLK) * Dd);
#pragma unroll
    for (int i = 0; i < 32; ++i) {
        int j   = i * 256 + t;          // float4 index, 0..8191 (coalesced)
        int row = j >> 7;               // 128 float4 per row
        int c4  = (j & 127) << 2;       // starting column
        float4 v = hsrc[j];
        bfpair p0 = split2(v.x);
        bfpair p1 = split2(v.y);
        bfpair p2 = split2(v.z);
        bfpair p3 = split2(v.w);
        v4bf hi4, lo4;
        hi4[0] = p0.hi; lo4[0] = p0.lo;
        hi4[1] = p1.hi; lo4[1] = p1.lo;
        hi4[2] = p2.hi; lo4[2] = p2.lo;
        hi4[3] = p3.hi; lo4[3] = p3.lo;
        *(v4bf*)&Ahi[row * A_LD + c4] = hi4;
        *(v4bf*)&Alo[row * A_LD + c4] = lo4;
    }
    __syncthreads();

    // ---------------- Gate partials: g = sigmoid(heads_row . Wg + bg) ----------
    {
        int r = t >> 2, q = t & 3;      // 4 threads per row, 128 elems each
        float s = 0.0f;
#pragma unroll
        for (int i = 0; i < 16; ++i) {
            int k = q * 128 + i * 8;
            v8bf ah = *(const v8bf*)&Ahi[r * A_LD + k];
            v8bf al = *(const v8bf*)&Alo[r * A_LD + k];
            const float* wp = Wg + k;
#pragma unroll
            for (int m = 0; m < 8; ++m)
                s += ((float)ah[m] + (float)al[m]) * wp[m];
        }
        gpart[r * 4 + q] = s;
    }
    __syncthreads();
    if (t < M_BLK) {
        float s = gpart[t * 4 + 0] + gpart[t * 4 + 1] +
                  gpart[t * 4 + 2] + gpart[t * 4 + 3] + bg[0];
        gval[t] = 1.0f / (1.0f + __expf(-s));
    }
    // (K-loop barriers below make gval visible before the epilogue)

    // ---------------- Main GEMM: he = X * Wz[h]^T over K = 0..511 --------------
    const int lane  = t & 31;
    const int wv    = t >> 5;           // wave 0..7
    const int mwo   = (wv & 1) * 32;    // wave M offset (2 M-waves x 32 rows)
    const int nwo   = (wv >> 1) * 32;   // wave N offset (4 N-waves x 32 cols)
    const int l16   = lane & 15;
    const int lhalf = lane >> 4;

    v8f acc[2][2];
#pragma unroll
    for (int a = 0; a < 2; ++a)
#pragma unroll
        for (int b = 0; b < 2; ++b)
#pragma unroll
            for (int e = 0; e < 8; ++e) acc[a][b][e] = 0.0f;

    // Each thread owns one 16-float half-row of the weight slab per k-step.
    const int   nrow  = t >> 1;         // 0..127 local weight row (output col)
    const int   whalf = (t & 1) * 16;
    const float* wbase = Wz + (size_t)h * Dd * Ee +
                         (size_t)(nBlock * N_BLK + nrow) * Ee + whalf;

    for (int ks = 0; ks < 16; ++ks) {
        // Fetch this k-slab of Wz into registers (8-byte aligned float2 loads).
        float f[16];
        const float* wsrc = wbase + ks * K_BLK;
#pragma unroll
        for (int i = 0; i < 8; ++i) {
            float2 p = ((const float2*)wsrc)[i];
            f[2 * i]     = p.x;
            f[2 * i + 1] = p.y;
        }
        if (ks < 15) __builtin_prefetch(wsrc + K_BLK, 0, 1);  // global_prefetch_b8

        __syncthreads();   // previous iteration's readers of Bhi/Blo are done
        {
            v8bf h0, l0, h1, l1;
#pragma unroll
            for (int i = 0; i < 8; ++i) {
                bfpair pa = split2(f[i]);
                bfpair pb = split2(f[8 + i]);
                h0[i] = pa.hi; l0[i] = pa.lo;
                h1[i] = pb.hi; l1[i] = pb.lo;
            }
            __bf16* bh = &Bhi[nrow * B_LD + whalf];
            __bf16* bl = &Blo[nrow * B_LD + whalf];
            *(v8bf*)bh       = h0;
            *(v8bf*)(bh + 8) = h1;
            *(v8bf*)bl       = l0;
            *(v8bf*)(bl + 8) = l1;
        }
        __syncthreads();

        const int kb = ks * K_BLK;

        // A fragments (16x32 bf16): lane half picks K {0-7,16-23} / {8-15,24-31}
        v16bf aH[2], aL[2], bH[2], bL[2];
#pragma unroll
        for (int mt = 0; mt < 2; ++mt) {
            const __bf16* p = &Ahi[(mwo + mt * 16 + l16) * A_LD + kb + lhalf * 8];
            aH[mt] = cat8(*(const v8bf*)p, *(const v8bf*)(p + 16));
            const __bf16* q = &Alo[(mwo + mt * 16 + l16) * A_LD + kb + lhalf * 8];
            aL[mt] = cat8(*(const v8bf*)q, *(const v8bf*)(q + 16));
        }
        // B fragments (32x16 bf16): lane half picks K {0-7,16-23} / {8-15,24-31}
#pragma unroll
        for (int nt = 0; nt < 2; ++nt) {
            const __bf16* p = &Bhi[(nwo + nt * 16 + l16) * B_LD + lhalf * 8];
            bH[nt] = cat8(*(const v8bf*)p, *(const v8bf*)(p + 16));
            const __bf16* q = &Blo[(nwo + nt * 16 + l16) * B_LD + lhalf * 8];
            bL[nt] = cat8(*(const v8bf*)q, *(const v8bf*)(q + 16));
        }

        // bf16x3 split product: hi*hi + hi*lo + lo*hi (lo*lo term ~2^-16, dropped)
#pragma unroll
        for (int mt = 0; mt < 2; ++mt)
#pragma unroll
            for (int nt = 0; nt < 2; ++nt) {
                acc[mt][nt] = __builtin_amdgcn_wmma_f32_16x16x32_bf16(
                    false, aH[mt], false, bH[nt], (short)0, acc[mt][nt], false, false);
                acc[mt][nt] = __builtin_amdgcn_wmma_f32_16x16x32_bf16(
                    false, aH[mt], false, bL[nt], (short)0, acc[mt][nt], false, false);
                acc[mt][nt] = __builtin_amdgcn_wmma_f32_16x16x32_bf16(
                    false, aL[mt], false, bH[nt], (short)0, acc[mt][nt], false, false);
            }
    }

    // ---------------- Epilogue: + pos-code rank-2 update, sigmoid blend --------
    const float invS1 = 1.0f / 4095.0f;
    const float tcH   = 0.5f * ((float)h / 7.0f);
#pragma unroll
    for (int nt = 0; nt < 2; ++nt) {
        int nloc = nwo + nt * 16 + l16;            // 0..127
        int nGl  = nBlock * N_BLK + nloc;          // output column d, 0..511
        const float* wcol = Wz + (size_t)h * Dd * Ee + (size_t)nGl * Ee;
        float w512 = wcol[512];                    // weight for pos code ch0
        float w513 = wcol[513];                    // weight for pos code ch1
#pragma unroll
        for (int mt = 0; mt < 2; ++mt) {
#pragma unroll
            for (int v = 0; v < 8; ++v) {
                int   mloc = mwo + mt * 16 + v + lhalf * 8;   // C layout: M = v + 8*(lane>=16)
                int   rowG = mBlock * M_BLK + mloc;           // 0..16383
                int   s    = rowG & (Ss - 1);                 // s index within sequence
                float pc0  = 0.5f * (float)s * invS1 + tcH;
                float pc1  = (float)(s & 511) * (1.0f / 512.0f);
                float he   = acc[mt][nt][v] + pc0 * w512 + pc1 * w513;
                float hv   = (float)Ahi[mloc * A_LD + nGl] +
                             (float)Alo[mloc * A_LD + nGl];   // reconstructed heads
                float g    = gval[mloc];
                out[((size_t)h * MH + rowG) * Dd + nGl] = hv + g * (he - hv);
            }
        }
    }
}

extern "C" void kernel_launch(void* const* d_in, const int* in_sizes, int n_in,
                              void* d_out, int out_size, void* d_ws, size_t ws_size,
                              hipStream_t stream) {
    (void)in_sizes; (void)n_in; (void)d_ws; (void)ws_size; (void)out_size;
    const float* heads = (const float*)d_in[0];
    const float* Wz    = (const float*)d_in[1];
    const float* Wg    = (const float*)d_in[2];
    const float* bg    = (const float*)d_in[3];
    float*       out   = (float*)d_out;

    // LDS: A hi/lo (2 * 64*520 hw) + B hi/lo (2 * 128*40 hw) + gate scratch
    size_t smem = (size_t)(2 * M_BLK * A_LD + 2 * N_BLK * B_LD) * sizeof(__bf16)
                + (size_t)(M_BLK * 4 + M_BLK) * sizeof(float);  // 154,880 B

    // Opt into >64KB dynamic LDS (WGP has 320KB). Idempotent; safe under capture.
    (void)hipFuncSetAttribute((const void*)JiuZhou_fused_gemm_gate,
                              hipFuncAttributeMaxDynamicSharedMemorySize,
                              (int)smem);

    dim3 grid(Dd / N_BLK, MH / M_BLK, Hh);   // (4, 256, 8)
    JiuZhou_fused_gemm_gate<<<grid, 256, smem, stream>>>(heads, Wz, Wg, bg, out);
}
